// GRU_49280454754768
// MI455X (gfx1250) — matrix operand order repaired and checked by reading, not compiled, not run on hardware
//
#include <hip/hip_runtime.h>
#include <hip/hip_bf16.h>

// GRU scan for MI455X (gfx1250, wave32, WMMA).
//
// Strategy:
//  - Batch rows are independent through the recurrence -> 4 workgroups of 16
//    batch rows each run the whole T=2048 scan as persistent kernels.
//  - Per step each WG computes  acc_i = x_t[16,256] @ Wi[256,768]  and
//    acc_h = h[16,256] @ Wh[256,768]  with v_wmma_f32_16x16x32_bf16
//    (bf16 inputs, f32 accumulate). 16 waves/WG; wave w owns gate-column
//    slab [16w,16w+16) of r, z and n (tiles n = w, w+16, w+32) so the gate
//    nonlinearity is entirely wave-local.
//  - Weights are pre-converted once into a fragment-major bf16 layout so
//    each B-operand fragment is one coalesced 32B-per-lane load. The weights
//    are L2-resident (768 KB total vs 192 MB L2) and are *streamed* every
//    step: an opaque per-iteration offset stops the compiler from hoisting
//    all 48 fragments into registers (which caused scratch spills).
//  - h is kept in LDS both as bf16 (WMMA A operand, row-major [m][k]) and
//    f32 (for the z*h term and output stores).

#define B_  64
#define T_  2048
#define F_  256
#define H_  256
#define G3_ 768            // 3*H
#define NT_ 48             // N tiles (768/16)
#define KB_ 8              // K blocks (256/32)
#define FRAG_ELEMS (NT_ * KB_ * 32 * 16)   // per-weight frag buffer, bf16 elems

typedef __bf16 bf16_t;
typedef __attribute__((ext_vector_type(16))) __bf16 v16bf;
typedef __attribute__((ext_vector_type(8)))  __bf16 v8bf;
typedef __attribute__((ext_vector_type(8)))  float  v8f;

__device__ __forceinline__ float sigmoid_(float v) {
    return 1.0f / (1.0f + __expf(-v));
}
__device__ __forceinline__ float tanh_(float v) {
    // stable: e->inf gives 1, e->0 gives -1
    float e = __expf(2.0f * v);
    return 1.0f - 2.0f / (e + 1.0f);
}

// ---------------------------------------------------------------------------
// Prologue: convert Wi/Wh (f32 [K=256][N=768] row-major) to bf16 fragment-
// major: frag[nt][kb32][lane][e] = W[k][n], n = nt*16 + (lane&15),
// k = kb32*32 + 16*(lane>>4) + e.  (contiguous k order, same as the A fragments.)
// ---------------------------------------------------------------------------
__global__ void gru_convert_weights(const float* __restrict__ Wi,
                                    const float* __restrict__ Wh,
                                    bf16_t* __restrict__ WiF,
                                    bf16_t* __restrict__ WhF) {
    int idx = blockIdx.x * blockDim.x + threadIdx.x;
    if (idx >= FRAG_ELEMS) return;
    int e    = idx & 15;
    int lane = (idx >> 4) & 31;
    int kb32 = (idx >> 9) & 7;
    int nt   = idx >> 12;
    int n = nt * 16 + (lane & 15);
    int k = kb32 * 32 + ((lane >> 4) << 4) + e;
    WiF[idx] = (bf16_t)Wi[k * G3_ + n];
    WhF[idx] = (bf16_t)Wh[k * G3_ + n];
}

// ---------------------------------------------------------------------------
// Persistent scan kernel: 4 blocks x 512 threads (16 waves).
// ---------------------------------------------------------------------------
__launch_bounds__(512, 1)
__global__ void gru_scan(const float* __restrict__ carry0,
                         const float* __restrict__ x,
                         const float* __restrict__ bh,
                         const bf16_t* __restrict__ WiF,
                         const bf16_t* __restrict__ WhF,
                         float* __restrict__ carry_out,
                         float* __restrict__ hidden_out) {
    __shared__ __align__(32) bf16_t sh_h[16 * 256];   // h as bf16 (A operand)
    __shared__ __align__(32) bf16_t sh_x[16 * 256];   // x_t as bf16 (A operand)
    __shared__ __align__(16) float  sh_hf[16 * 256];  // h as f32 (epilogue)

    const int tid  = threadIdx.x;
    const int lane = tid & 31;
    const int w    = tid >> 5;       // wave id 0..15
    const int wg   = blockIdx.x;     // batch-row group 0..3
    const int mrow = lane & 15;      // A-row / B-col / C-col role of this lane
    const int g    = lane >> 4;      // lane half-group
    const int col  = w * 16 + mrow;  // this lane's gate column (0..255)

    // --- init h from carry (f32 + bf16 copies in LDS) ---
    {
        int m  = tid >> 5;
        int k0 = (tid & 31) * 8;
        const float* src = carry0 + (wg * 16 + m) * H_ + k0;
        #pragma unroll
        for (int j = 0; j < 8; ++j) {
            float v = src[j];
            sh_hf[m * 256 + k0 + j] = v;
            sh_h [m * 256 + k0 + j] = (bf16_t)v;
        }
    }

    // biases are time-invariant: load once per lane
    const float bias_r = bh[        col];
    const float bias_z = bh[256 +   col];
    const float bias_n = bh[512 +   col];

    __syncthreads();

    for (int t = 0; t < T_; ++t) {
        // Opaque zero offset: prevents the compiler from proving the weight
        // fragments loop-invariant over t (which hoists 384 VGPRs of weights
        // out of the scan and spills them to scratch). Forces clean per-step
        // streaming of the L2-resident fragments instead.
        long opq = 0;
        asm volatile("" : "+s"(opq));
        const bf16_t* __restrict__ WiFt = WiF + opq;
        const bf16_t* __restrict__ WhFt = WhF + opq;

        // --- stage x[:, t, :] into LDS as bf16 ---
        {
            int m  = tid >> 5;
            int k0 = (tid & 31) * 8;
            const float* src = x + (((size_t)(wg * 16 + m)) * T_ + t) * F_ + k0;
            #pragma unroll
            for (int j = 0; j < 8; ++j)
                sh_x[m * 256 + k0 + j] = (bf16_t)src[j];
            if (t + 1 < T_)
                __builtin_prefetch(src + F_, 0, 3);  // global_prefetch_b8 (near)
        }
        __syncthreads();

        // --- K loop: 8 blocks of K=32, 6 WMMAs each ---
        v8f acc_ir{}, acc_iz{}, acc_in{};
        v8f acc_hr{}, acc_hz{}, acc_hn{};

        #pragma unroll
        for (int kb32 = 0; kb32 < KB_; ++kb32) {
            const int kb = kb32 * 32;

            // A fragments from LDS (contiguous k = kb + 16*g + s, as in the B frags)
            v16bf a_x, a_h;
            {
                const bf16_t* px = &sh_x[mrow * 256 + kb + 16 * g];
                const bf16_t* ph = &sh_h[mrow * 256 + kb + 16 * g];
                v8bf xlo = *(const v8bf*)(px);
                v8bf xhi = *(const v8bf*)(px + 8);
                v8bf hlo = *(const v8bf*)(ph);
                v8bf hhi = *(const v8bf*)(ph + 8);
                #pragma unroll
                for (int j = 0; j < 8; ++j) {
                    a_x[j]     = xlo[j];  a_x[8 + j] = xhi[j];
                    a_h[j]     = hlo[j];  a_h[8 + j] = hhi[j];
                }
            }

            // 3 gate tiles for this wave: nt = w (r), w+16 (z), w+32 (n)
            {
                const int nt = w;
                const size_t fb = (((size_t)nt * KB_ + kb32) * 32 + lane) * 16;
                const v16bf bi = *(const v16bf*)&WiFt[fb];
                const v16bf bw = *(const v16bf*)&WhFt[fb];
                acc_ir = __builtin_amdgcn_wmma_f32_16x16x32_bf16(
                    false, a_x, false, bi, (short)0, acc_ir, false, false);
                acc_hr = __builtin_amdgcn_wmma_f32_16x16x32_bf16(
                    false, a_h, false, bw, (short)0, acc_hr, false, false);
            }
            {
                const int nt = w + 16;
                const size_t fb = (((size_t)nt * KB_ + kb32) * 32 + lane) * 16;
                const v16bf bi = *(const v16bf*)&WiFt[fb];
                const v16bf bw = *(const v16bf*)&WhFt[fb];
                acc_iz = __builtin_amdgcn_wmma_f32_16x16x32_bf16(
                    false, a_x, false, bi, (short)0, acc_iz, false, false);
                acc_hz = __builtin_amdgcn_wmma_f32_16x16x32_bf16(
                    false, a_h, false, bw, (short)0, acc_hz, false, false);
            }
            {
                const int nt = w + 32;
                const size_t fb = (((size_t)nt * KB_ + kb32) * 32 + lane) * 16;
                const v16bf bi = *(const v16bf*)&WiFt[fb];
                const v16bf bw = *(const v16bf*)&WhFt[fb];
                acc_in = __builtin_amdgcn_wmma_f32_16x16x32_bf16(
                    false, a_x, false, bi, (short)0, acc_in, false, false);
                acc_hn = __builtin_amdgcn_wmma_f32_16x16x32_bf16(
                    false, a_h, false, bw, (short)0, acc_hn, false, false);
            }
        }

        __syncthreads();   // all reads of sh_h / sh_x done

        // --- gates + h update. C layout: elem r, lane -> (M = r+8g, N = mrow)
        #pragma unroll
        for (int r = 0; r < 8; ++r) {
            const int m    = r + 8 * g;
            const float ho = sh_hf[m * 256 + col];
            const float rg = sigmoid_(acc_ir[r] + acc_hr[r] + bias_r);
            const float zg = sigmoid_(acc_iz[r] + acc_hz[r] + bias_z);
            const float ng = tanh_(acc_in[r] + rg * (acc_hn[r] + bias_n));
            const float hn = (1.0f - zg) * ng + zg * ho;

            const int b = wg * 16 + m;
            hidden_out[(((size_t)b) * T_ + t) * H_ + col] = hn;
            sh_hf[m * 256 + col] = hn;
            sh_h [m * 256 + col] = (bf16_t)hn;
            if (t == T_ - 1)
                carry_out[b * H_ + col] = hn;
        }
        __syncthreads();   // h update visible before next step's reads
    }
}

// ---------------------------------------------------------------------------
extern "C" void kernel_launch(void* const* d_in, const int* in_sizes, int n_in,
                              void* d_out, int out_size, void* d_ws, size_t ws_size,
                              hipStream_t stream) {
    const float* carry = (const float*)d_in[0];   // [B,H]
    const float* x     = (const float*)d_in[1];   // [B,T,F]
    const float* Wi    = (const float*)d_in[2];   // [F,3H]
    const float* Wh    = (const float*)d_in[3];   // [H,3H]
    const float* bh    = (const float*)d_in[4];   // [3H]

    float* out        = (float*)d_out;
    float* carry_out  = out;              // [B,H] first in return order
    float* hidden_out = out + B_ * H_;    // then [B,T,H]

    bf16_t* WiF = (bf16_t*)d_ws;                  // 384 KB
    bf16_t* WhF = WiF + FRAG_ELEMS;               // 384 KB

    const int total = FRAG_ELEMS;
    gru_convert_weights<<<(total + 255) / 256, 256, 0, stream>>>(Wi, Wh, WiF, WhF);
    gru_scan<<<B_ / 16, 512, 0, stream>>>(carry, x, bh, WiF, WhF,
                                          carry_out, hidden_out);
}
